// LabelAttention_32384053412296
// MI455X (gfx1250) — matrix operand order repaired and checked by reading, not compiled, not run on hardware
//
#include <hip/hip_runtime.h>

// CDNA5 / gfx1250 WMMA types
typedef __attribute__((ext_vector_type(16))) __bf16 v16bf;
typedef __attribute__((ext_vector_type(8)))  float  v8f;
typedef __attribute__((ext_vector_type(4)))  float  f32x4;

#define D_DIM  1024
#define UNITS  1024
#define NROWS  65536          // B*T = 16*4096
#define SCALE  0.03125f       // 1/sqrt(1024)

// ---------------------------------------------------------------------------
// Kernel 1: pack = bf16( scale * W^T W ) stored in WMMA-B-fragment layout.
//   M[k,d] = scale * sum_u W[u,k] * W[u,d]
// Packed layout: block (kb,nb) covers K rows [kb*32,kb*32+32) x N cols
// [nb*16,nb*16+16). Each block is 512 bf16 (1 KB): offset = lane*16 + elem,
// lane = (((k%32)/8)%2==0 ? n%16 : 16+n%16), elem = k%8 + (k%32>=16 ? 8 : 0)  -> matches the ISA's
// 16-bit 32x16 B-matrix VGPR striping, so kernel 2 loads it contiguously.
// ---------------------------------------------------------------------------
__global__ __launch_bounds__(256) void k_wtw_pack(const float* __restrict__ W,
                                                  __bf16* __restrict__ pack) {
  const int lane = threadIdx.x & 31;
  const int wave = threadIdx.x >> 5;
  const int tile = blockIdx.x * 8 + wave;      // 0..4095
  const int k0   = (tile >> 6) * 16;           // M-row tile (K dim of final GEMM)
  const int n0   = (tile & 63) * 16;           // N-col tile
  const int half = lane >> 4;
  const int lid  = lane & 15;

  const int acol = k0 + lid;
  const int a_u0 = half ? 16 : 0;              // A: 0..15 / 16..31
  const int bcol = n0 + lid;
  const int b_u0 = half ? 16 : 0;              // B: 0..15 / 16..31

  v8f acc = {};
  for (int u0 = 0; u0 < UNITS; u0 += 32) {
    v16bf a, b;
#pragma unroll
    for (int e = 0; e < 8; ++e) {
      a[e]     = (__bf16)W[(u0 + a_u0 + e) * D_DIM + acol];
      a[e + 8] = (__bf16)W[(u0 + a_u0 + 8 + e) * D_DIM + acol];
    }
#pragma unroll
    for (int e = 0; e < 16; ++e) {
      b[e] = (__bf16)W[(u0 + b_u0 + e) * D_DIM + bcol];
    }
    acc = __builtin_amdgcn_wmma_f32_16x16x32_bf16(false, a, false, b,
                                                  (short)0, acc, false, false);
  }

#pragma unroll
  for (int r = 0; r < 8; ++r) {
    const int krow  = k0 + (half ? (8 + r) : r);
    const int block = (krow >> 5) * 64 + (n0 >> 4);
    const int lsel  = ((krow >> 3) & 1) ? (lid + 16) : lid;
    pack[block * 512 + lsel * 16 + ((krow & 7) + ((krow & 16) ? 8 : 0))] = (__bf16)(acc[r] * SCALE);
  }
}

// ---------------------------------------------------------------------------
// A-fragment loader: X[row, k0 + {g0..g0+3, g0+4..g0+7, g0+16.., g0+20..}]
// (g0 = 0 for lanes 0-15, 8 for lanes 16-31), f32 -> bf16.
// ---------------------------------------------------------------------------
__device__ __forceinline__ v16bf load_a_frag(const float* __restrict__ xr, int g0) {
  f32x4 a0 = *(const f32x4*)(xr + g0);
  f32x4 a1 = *(const f32x4*)(xr + g0 + 4);
  f32x4 a2 = *(const f32x4*)(xr + g0 + 16);
  f32x4 a3 = *(const f32x4*)(xr + g0 + 20);
  v16bf a;
#pragma unroll
  for (int e = 0; e < 4; ++e) {
    a[e]      = (__bf16)a0[e];
    a[e + 4]  = (__bf16)a1[e];
    a[e + 8]  = (__bf16)a2[e];
    a[e + 12] = (__bf16)a3[e];
  }
  return a;
}

// ---------------------------------------------------------------------------
// Kernel 2: out = X @ M.
// Workgroup tile = 128 rows x 256 cols; 8 waves = 4 row-groups x 2 col-groups,
// each wave 32 rows x 128 cols (2 A frags x 8 B frags -> 16 acc tiles).
// B fragments are re-read by the 4 row-group waves at identical addresses
// (WGP$ hits), so unique B traffic from L2 is 512 KB per workgroup -> ~1 GB
// total instead of 8 GB with a 16-row workgroup. blockIdx.x = column block
// (fast-varying) so the 4 blocks sharing a 512 KB X row-panel co-schedule and
// X streams from HBM ~once through L2.
// ---------------------------------------------------------------------------
__global__ __launch_bounds__(256) void k_xm_gemm(const float* __restrict__ X,
                                                 const __bf16* __restrict__ pack,
                                                 float* __restrict__ out) {
  const int lane  = threadIdx.x & 31;
  const int wave  = threadIdx.x >> 5;
  const int cg    = wave & 1;                  // col-group 0..1
  const int rg    = wave >> 1;                 // row-group 0..3
  const int row0  = blockIdx.y * 128 + rg * 32;
  const int nbase = blockIdx.x * 256 + cg * 128;
  const int half  = lane >> 4;
  const int lid   = lane & 15;
  const int g0    = half ? 8 : 0;

  const float* xr0 = X + (size_t)(row0 + lid) * D_DIM;        // A frag 0 rows
  const float* xr1 = X + (size_t)(row0 + 16 + lid) * D_DIM;   // A frag 1 rows

  v8f acc[2][8] = {};

  for (int k0 = 0; k0 < D_DIM; k0 += 32) {
    // CDNA5 global_prefetch_b8: pull next K-step's X lines toward the WGP
    __builtin_prefetch(xr0 + k0 + 32, 0, 1);
    __builtin_prefetch(xr1 + k0 + 32, 0, 1);

    const v16bf a0 = load_a_frag(xr0 + k0, g0);
    const v16bf a1 = load_a_frag(xr1 + k0, g0);

    const __bf16* bp =
        pack + ((size_t)(k0 >> 5) * 64 + (nbase >> 4)) * 512 + lane * 16;
#pragma unroll
    for (int t = 0; t < 8; ++t) {
      const v16bf b = *(const v16bf*)(bp + t * 512);   // contiguous 32 B/lane
      acc[0][t] = __builtin_amdgcn_wmma_f32_16x16x32_bf16(false, a0, false, b,
                                                          (short)0, acc[0][t],
                                                          false, false);
      acc[1][t] = __builtin_amdgcn_wmma_f32_16x16x32_bf16(false, a1, false, b,
                                                          (short)0, acc[1][t],
                                                          false, false);
    }
  }

  // Epilogue: D layout (VGPR r -> row r / 8+r by half, lane -> col).
#pragma unroll
  for (int g = 0; g < 2; ++g) {
    const int rbase = row0 + g * 16 + (half ? 8 : 0);
#pragma unroll
    for (int t = 0; t < 8; ++t) {
      const int col = nbase + t * 16 + lid;
#pragma unroll
      for (int r = 0; r < 8; ++r) {
        out[(size_t)(rbase + r) * D_DIM + col] = acc[g][t][r];
      }
    }
  }
}

// ---------------------------------------------------------------------------
extern "C" void kernel_launch(void* const* d_in, const int* in_sizes, int n_in,
                              void* d_out, int out_size, void* d_ws, size_t ws_size,
                              hipStream_t stream) {
  (void)in_sizes; (void)n_in; (void)out_size; (void)ws_size;
  const float* X = (const float*)d_in[0];      // (16,4096,1024) f32
  const float* W = (const float*)d_in[1];      // (1024,1024)    f32
  float*     out = (float*)d_out;              // (16,4096,1024) f32
  __bf16*   pack = (__bf16*)d_ws;              // 2 MB packed bf16 M = scale*W^T W

  // Kernel 1: 4096 tiles / 8 waves per block -> 512 blocks
  k_wtw_pack<<<512, 256, 0, stream>>>(W, pack);
  // Kernel 2: grid = (N/256 col-blocks, M/128 row-blocks)
  dim3 grid2(D_DIM / 256, NROWS / 128);
  k_xm_gemm<<<grid2, 256, 0, stream>>>(X, pack, out);
}
